// TensorSketchLayer_84456236909171
// MI455X (gfx1250) — matrix operand order repaired and checked by reading, not compiled, not run on hardware
//
#include <hip/hip_runtime.h>

// TensorSketch on MI455X (gfx1250):
//   K0a: hmaps f32 -> f16 copy (hmH), K0b: x f32 -> f16 transpose (xT[n][k]).
//   K1 : count[o] = hmaps[o] @ x via v_wmma_f32_16x16x32_f16, tiles staged by
//        the Tensor Data Mover (tensor_load_to_lds, double buffered, LDS row
//        padding done by the TDM pad engine), result stored [o][n][d] f16.
//   K2 : per column n: FFT(8192) x3 -> complex product -> IFFT -> real -> out.
//        LDS-resident radix-4 Stockham (6 radix-4 + 1 radix-2), 192 KB LDS.

typedef __attribute__((ext_vector_type(16))) _Float16 v16h;
typedef __attribute__((ext_vector_type(8)))  _Float16 v8h;
typedef __attribute__((ext_vector_type(8)))  float    v8f;
typedef __attribute__((ext_vector_type(4)))  unsigned int v4u;
typedef __attribute__((ext_vector_type(8)))  int      v8i;
typedef __attribute__((ext_vector_type(4)))  int      v4i;

#define D_IN   1024
#define D_OUT  8192
#define NCOL   2048
#define NORDER 3

#define BM 128
#define BN 128
#define BK 32
#define LDA 34   // halves per LDS row: 32 data + 1 DWORD TDM pad -> 17-bank stride
#define LDB 34

// ---------------------------------------------------------------------------
// K0a: hmaps f32 -> f16 (straight convert, coalesced)
// ---------------------------------------------------------------------------
__global__ void ts_convert_hm(const float* __restrict__ hm, _Float16* __restrict__ hmH)
{
  size_t total = (size_t)NORDER * D_OUT * D_IN;
  for (size_t i = (size_t)blockIdx.x * blockDim.x + threadIdx.x; i < total;
       i += (size_t)gridDim.x * blockDim.x)
    hmH[i] = (_Float16)hm[i];
}

// ---------------------------------------------------------------------------
// K0b: x[k][n] f32 -> xT[n][k] f16 (LDS-tiled 32x32 transpose)
// ---------------------------------------------------------------------------
__global__ void ts_xpose(const float* __restrict__ x, _Float16* __restrict__ xT)
{
  __shared__ float t[32][33];
  const int n0 = blockIdx.x * 32, k0 = blockIdx.y * 32;
  const int tx = threadIdx.x, ty = threadIdx.y;       // 32 x 8
  for (int r = ty; r < 32; r += 8)
    t[r][tx] = x[(size_t)(k0 + r) * NCOL + n0 + tx];
  __syncthreads();
  for (int r = ty; r < 32; r += 8)
    xT[(size_t)(n0 + r) * D_IN + k0 + tx] = (_Float16)t[tx][r];
}

// ---------------------------------------------------------------------------
// TDM helper: issue a 2-D tensor_load_to_lds of a tile_d1 x tile_d0 f16 tile
// (row stride `stride0` elements) into LDS at byte offset lds_off, with the
// TDM pad engine inserting 1 DWORD after every 16 DWORDs (64B = one 32-half
// row) -> LDS row pitch of 34 halves. (CDNA5 ISA)
// ---------------------------------------------------------------------------
__device__ __forceinline__ unsigned lds_off_of(const void* p)
{
  return (unsigned)(size_t)(__attribute__((address_space(3))) const void*)p;
}

__device__ __forceinline__ void tdm_load_tile(unsigned lds_off,
                                              const _Float16* gptr,
                                              unsigned tensor_d0, unsigned tensor_d1,
                                              unsigned stride0,
                                              unsigned tile_d0, unsigned tile_d1)
{
  unsigned long long ga = (unsigned long long)(size_t)gptr;
  v4u g0;
  g0[0] = 1u;                                          // count=1, user descriptor
  g0[1] = lds_off;                                     // lds_addr (bytes)
  g0[2] = (unsigned)(ga & 0xffffffffu);                // global_addr[31:0]
  g0[3] = (unsigned)((ga >> 32) & 0x01ffffffu) | (2u << 30);  // addr[56:32] | type=2
  v8i g1;
  g1[0] = (int)((1u << 16) |                           // data_size = 2 bytes
                (1u << 20) |                           // pad_enable
                (3u << 22));                           // pad_interval: 16 DWORDs
                                                       // pad_amount code 0 = 1 DWORD
  g1[1] = (int)((tensor_d0 & 0xffffu) << 16);          // abar_addr=0 | tensor_dim0.lo
  g1[2] = (int)(((tensor_d0 >> 16) & 0xffffu) | ((tensor_d1 & 0xffffu) << 16));
  g1[3] = (int)(((tensor_d1 >> 16) & 0xffffu) | ((tile_d0 & 0xffffu) << 16));
  g1[4] = (int)(tile_d1 & 0xffffu);                    // tile_dim1 | tile_dim2=0
  g1[5] = (int)stride0;                                // tensor_dim0_stride[31:0]
  g1[6] = 0;                                           // stride0[47:32] | td1_stride.lo
  g1[7] = 0;
  v4i g2 = {0, 0, 0, 0};
  v4i g3 = {0, 0, 0, 0};
#if __clang_major__ >= 23
  v8i gz = {0, 0, 0, 0, 0, 0, 0, 0};
  __builtin_amdgcn_tensor_load_to_lds(g0, g1, g2, g3, gz, 0);
#else
  __builtin_amdgcn_tensor_load_to_lds(g0, g1, g2, g3, 0);
#endif
}

// ---------------------------------------------------------------------------
// K1: WMMA GEMM  countT[o][n][d] = sum_k hm[o][d][k] * x[k][n]
//   128x128 block tile, K-step 32, 8 waves (2x4), wave tile 64x32 = 4x2 WMMAs,
//   A/B tiles DMA'd by the TDM, double buffered.
// ---------------------------------------------------------------------------
__global__ __launch_bounds__(256) void ts_count_wmma(
    const _Float16* __restrict__ xT, const _Float16* __restrict__ hmH,
    _Float16* __restrict__ countT)
{
  __shared__ __align__(16) _Float16 As[2][BM * LDA];   // [m][k], TDM-padded rows
  __shared__ __align__(16) _Float16 Bst[2][BN * LDB];  // [n][k], TDM-padded rows

  const int tid  = threadIdx.x;
  const int lane = tid & 31;
  const int wave = tid >> 5;       // 0..7
  const int wm   = wave >> 2;      // 0..1 -> 64 rows of d
  const int wn   = wave & 3;       // 0..3 -> 32 cols of n
  const int lm   = lane & 15;
  const int hi   = lane >> 4;

  const int o     = blockIdx.z;
  const int dbase = blockIdx.y * BM;
  const int nbase = blockIdx.x * BN;

  const _Float16* __restrict__ Abase = hmH + ((size_t)o * D_OUT + dbase) * D_IN;
  const _Float16* __restrict__ Bbase = xT + (size_t)nbase * D_IN;

  v8f acc[4][2] = {};
  const int NT = D_IN / BK;

  if (wave == 0) {   // single TDM issuer per workgroup (EXEC-ignored wave op)
    tdm_load_tile(lds_off_of(As[0]),  Abase, D_IN, D_OUT, D_IN, BK, BM);
    tdm_load_tile(lds_off_of(Bst[0]), Bbase, D_IN, NCOL,  D_IN, BK, BN);
  }

  for (int kt = 0; kt < NT; ++kt) {
    if (wave == 0) {
      if (kt + 1 < NT) {
        const int kb = (kt + 1) * BK;
        const int nb = (kt + 1) & 1;
        tdm_load_tile(lds_off_of(As[nb]),  Abase + kb, D_IN, D_OUT, D_IN, BK, BM);
        tdm_load_tile(lds_off_of(Bst[nb]), Bbase + kb, D_IN, NCOL,  D_IN, BK, BN);
        __builtin_amdgcn_s_wait_tensorcnt((short)2);  // current tile's pair done
      } else {
        __builtin_amdgcn_s_wait_tensorcnt((short)0);
      }
    }
    __syncthreads();

    const int cb = kt & 1;
    const uint32_t* as32 = (const uint32_t*)As[cb];
    const uint32_t* bs32 = (const uint32_t*)Bst[cb];

    // A fragment (16x32 f16): lane L holds M=L&15; slot v covers
    //   K = 2*(v&3) + 8*(L>>4) + 16*(v>>2) .. +1   (ISA 05_wmma layout)
    union { v16h h; uint32_t u[8]; } af[4];
    for (int sm = 0; sm < 4; ++sm) {
      int row = wm * 64 + sm * 16 + lm;
      #pragma unroll
      for (int v = 0; v < 8; ++v) {
        int k0 = 2 * (v & 3) + 8 * hi + 16 * (v >> 2);
        af[sm].u[v] = as32[(row * LDA + k0) >> 1];
      }
    }
    // B fragment (32x16 f16): lane L holds N=L&15; slot v covers K = 2*(v&3) + 8*(L>>4) + 16*(v>>2)
    union { v16h h; uint32_t u[8]; } bf[2];
    for (int sn = 0; sn < 2; ++sn) {
      int row = wn * 32 + sn * 16 + lm;
      #pragma unroll
      for (int v = 0; v < 8; ++v) {
        int k0 = 2 * (v & 3) + 8 * hi + 16 * (v >> 2);
        bf[sn].u[v] = bs32[(row * LDB + k0) >> 1];
      }
    }

    #pragma unroll
    for (int sm = 0; sm < 4; ++sm)
      #pragma unroll
      for (int sn = 0; sn < 2; ++sn)
        acc[sm][sn] = __builtin_amdgcn_wmma_f32_16x16x32_f16(
            false, af[sm].h, false, bf[sn].h, (short)0, acc[sm][sn], false, false);

    __syncthreads();
  }

  // Write back transposed: countT[o][n][d] (f16). For a lane, acc row r maps to
  // d = base + 8*hi + r -> 8 consecutive d's -> one 16B store.
  for (int sm = 0; sm < 4; ++sm)
    for (int sn = 0; sn < 2; ++sn) {
      int n  = nbase + wn * 32 + sn * 16 + lm;
      int d0 = dbase + wm * 64 + sm * 16 + 8 * hi;
      v8h hv;
      #pragma unroll
      for (int r = 0; r < 8; ++r) hv[r] = (_Float16)acc[sm][sn][r];
      *(v8h*)(countT + ((size_t)o * NCOL + n) * D_OUT + d0) = hv;
    }
}

// ---------------------------------------------------------------------------
// K2: per-column FFT pipeline in LDS.
// Stockham DIF, N=8192 = 4^6 * 2: six radix-4 passes + one radix-2 pass.
// Natural-order in/out; ping-pongs x<->y; result lands in y.
// ---------------------------------------------------------------------------
__device__ __forceinline__ void fft8192(float* xr, float* xi, float* yr, float* yi,
                                        float sgn, int tid, int nthr)
{
  #pragma unroll
  for (int pass = 0; pass < 6; ++pass) {
    const int logm = 2 * pass;
    const int m    = 1 << logm;
    const int l    = 2048 >> logm;
    const float tw = sgn * 6.28318530717958647f / (float)(4 * l);
    for (int idx = tid; idx < 2048; idx += nthr) {
      const int k    = idx & (m - 1);
      const int j    = idx >> logm;
      const int base = k + (j << logm);           // j*m + k  (m*l == N/4 always)
      float c0r = xr[base],        c0i = xi[base];
      float c1r = xr[base + 2048], c1i = xi[base + 2048];
      float c2r = xr[base + 4096], c2i = xi[base + 4096];
      float c3r = xr[base + 6144], c3i = xi[base + 6144];
      float t0r = c0r + c2r, t0i = c0i + c2i;
      float t1r = c0r - c2r, t1i = c0i - c2i;
      float t2r = c1r + c3r, t2i = c1i + c3i;
      float d1r = c1r - c3r, d1i = c1i - c3i;
      float t3r = -sgn * d1i, t3i = sgn * d1r;    // i*sgn*(c1-c3)
      float s1, cw;
      __sincosf(tw * (float)j, &s1, &cw);
      float w1r = cw, w1i = s1;
      float w2r = w1r * w1r - w1i * w1i, w2i = 2.0f * w1r * w1i;
      float w3r = w2r * w1r - w2i * w1i, w3i = w2r * w1i + w2i * w1r;
      const int ob = k + (j << (logm + 2));       // k + 4*j*m
      yr[ob] = t0r + t2r;                 yi[ob] = t0i + t2i;
      float u1r = t1r + t3r, u1i = t1i + t3i;
      yr[ob + m]     = w1r * u1r - w1i * u1i;  yi[ob + m]     = w1r * u1i + w1i * u1r;
      float u2r = t0r - t2r, u2i = t0i - t2i;
      yr[ob + 2 * m] = w2r * u2r - w2i * u2i;  yi[ob + 2 * m] = w2r * u2i + w2i * u2r;
      float u3r = t1r - t3r, u3i = t1i - t3i;
      yr[ob + 3 * m] = w3r * u3r - w3i * u3i;  yi[ob + 3 * m] = w3r * u3i + w3i * u3r;
    }
    __syncthreads();
    float* t;
    t = xr; xr = yr; yr = t;
    t = xi; xi = yi; yi = t;
  }
  // Final radix-2 (l=1, m=4096, unit twiddle). After 6 swaps x is the original
  // input buffer; result goes to y (= caller's y buffers).
  for (int idx = tid; idx < 4096; idx += nthr) {
    float ar = xr[idx],        ai = xi[idx];
    float br = xr[idx + 4096], bi = xi[idx + 4096];
    yr[idx]        = ar + br;  yi[idx]        = ai + bi;
    yr[idx + 4096] = ar - br;  yi[idx + 4096] = ai - bi;
  }
  __syncthreads();
}

__global__ __launch_bounds__(512) void ts_fft_kernel(
    const _Float16* __restrict__ countT, float* __restrict__ out)
{
  extern __shared__ float lds[];          // 6 * 8192 floats = 192 KB
  float* Ar = lds;
  float* Ai = lds + 1 * 8192;
  float* Br = lds + 2 * 8192;
  float* Bi = lds + 3 * 8192;
  float* Pr = lds + 4 * 8192;
  float* Pi = lds + 5 * 8192;

  const int n   = blockIdx.x;
  const int tid = threadIdx.x;

  for (int o = 0; o < NORDER; ++o) {
    const _Float16* __restrict__ src = countT + ((size_t)o * NCOL + n) * D_OUT;
    for (int d = tid; d < D_OUT; d += 512) { Ar[d] = (float)src[d]; Ai[d] = 0.0f; }
    __syncthreads();
    fft8192(Ar, Ai, Br, Bi, -1.0f, tid, 512);   // forward; result in B
    if (o == 0) {
      for (int d = tid; d < D_OUT; d += 512) { Pr[d] = Br[d]; Pi[d] = Bi[d]; }
    } else {
      for (int d = tid; d < D_OUT; d += 512) {
        float pr = Pr[d], pi = Pi[d], br = Br[d], bi = Bi[d];
        Pr[d] = pr * br - pi * bi;
        Pi[d] = pr * bi + pi * br;
      }
    }
    __syncthreads();
  }

  fft8192(Pr, Pi, Br, Bi, +1.0f, tid, 512);     // inverse; result in B
  const float scale = 1.0f / 8192.0f;
  for (int d = tid; d < D_OUT; d += 512)
    out[(size_t)d * NCOL + n] = Br[d] * scale;  // out is [D_OUT, NCOL]
}

// ---------------------------------------------------------------------------
extern "C" void kernel_launch(void* const* d_in, const int* in_sizes, int n_in,
                              void* d_out, int out_size, void* d_ws, size_t ws_size,
                              hipStream_t stream)
{
  const float* x  = (const float*)d_in[0];   // [D_IN, NCOL] f32
  const float* hm = (const float*)d_in[1];   // [NORDER, D_OUT, D_IN] f32
  float* out = (float*)d_out;                // [D_OUT, NCOL] f32

  // Workspace layout: countT (96 MB) | hmH (48 MB) | xT (4 MB)
  _Float16* countT = (_Float16*)d_ws;
  _Float16* hmH = (_Float16*)((char*)d_ws + (size_t)NORDER * NCOL * D_OUT * 2);
  _Float16* xT  = (_Float16*)((char*)hmH + (size_t)NORDER * D_OUT * D_IN * 2);

  ts_convert_hm<<<4096, 256, 0, stream>>>(hm, hmH);

  dim3 gt(NCOL / 32, D_IN / 32);
  ts_xpose<<<gt, dim3(32, 8), 0, stream>>>(x, xT);

  dim3 g1(NCOL / BN, D_OUT / BM, NORDER);    // (16, 64, 3)
  ts_count_wmma<<<g1, 256, 0, stream>>>(xT, hmH, countT);

  ts_fft_kernel<<<NCOL, 512, 6 * D_OUT * sizeof(float), stream>>>(countT, out);
}
